// WindowAttention_50577534877903
// MI455X (gfx1250) — matrix operand order repaired and checked by reading, not compiled, not run on hardware
//
#include <hip/hip_runtime.h>
#include <hip/hip_bf16.h>
#include <math.h>

typedef _Float16 half_t;
typedef __attribute__((ext_vector_type(16))) _Float16 v16h;
typedef __attribute__((ext_vector_type(8)))  _Float16 v8h;
typedef __attribute__((ext_vector_type(8)))  float    v8f;

#define DIM   128
#define NTOK  64
#define HEADS 4
#define HD    32

union V16H { v16h v; v8h h[2]; };

// ---- WMMA fragment helpers (wave32, 16x16x32 f16 -> f32) -------------------
// A 16x32 (MxK), row-major source with leading dim ld (halves).
// Lane layout (ISA 7.12.2): lanes 0-15 -> M=lane, K = 0..7 & 16..23;
//                           lanes 16-31 -> M=lane-16, K = 8..15 & 24..31.
__device__ inline v16h load_frag_a(const half_t* base, int ld, int lane) {
    int m  = lane & 15;
    int kb = (lane & 16) ? 8 : 0;
    const half_t* p = base + m * ld + kb;
    V16H u;
    u.h[0] = *(const v8h*)(p);
    u.h[1] = *(const v8h*)(p + 16);
    return u.v;
}

// B 32x16 (KxN) from N-major ("transposed") source bt[n][k], leading dim ld.
// Lane holds column N=lane&15; lanes 0-15 K=0..7 & 16..23, lanes 16-31 K=8..15 & 24..31.
__device__ inline v16h load_frag_b(const half_t* bt, int ld, int lane) {
    int n  = lane & 15;
    int kb = (lane & 16) ? 8 : 0;
    const half_t* p = bt + n * ld + kb;
    V16H u;
    u.h[0] = *(const v8h*)(p);
    u.h[1] = *(const v8h*)(p + 16);
    return u.v;
}

__device__ inline v8f wmma_f16(v16h a, v16h b, v8f c) {
    return __builtin_amdgcn_wmma_f32_16x16x32_f16(
        /*neg_a=*/false, a, /*neg_b=*/false, b,
        /*c_mod=*/(short)0, c, /*reuse_a=*/false, /*reuse_b=*/false);
}

__device__ inline float sgn(float t) { return (t > 0.f) ? 1.f : ((t < 0.f) ? -1.f : 0.f); }

// ---- Kernel 0: CPB-MLP bias table + weight transpose/convert ---------------
__global__ void swin_prep_kernel(const float* __restrict__ W_qkv,
                                 const float* __restrict__ W_proj,
                                 const float* __restrict__ cpb_w1,
                                 const float* __restrict__ cpb_b1,
                                 const float* __restrict__ cpb_w2,
                                 const float* __restrict__ h_times,
                                 const float* __restrict__ pow_para,
                                 const float* __restrict__ ws1,
                                 const float* __restrict__ ws2,
                                 const int*   __restrict__ rel_index,
                                 const float* __restrict__ coords,
                                 float* __restrict__ biasg,     // [4][64][64] f32
                                 half_t* __restrict__ WqkvT,    // [384][128] f16 (N-major)
                                 half_t* __restrict__ WprojT) { // [128][128] f16 (N-major)
    __shared__ float table[225][HEADS];
    const int tid = threadIdx.x;

    const float ht = log1pf(__expf(h_times[0]));                  // softplus
    const float pp = 1.f / (1.f + __expf(-pow_para[0]));          // sigmoid
    const float ws = log1pf(__expf(ws1[0])) / log1pf(__expf(ws2[0]));

    if (tid < 225) {
        float t0 = coords[tid * 2 + 0] * ht;
        float t1 = coords[tid * 2 + 1] * ht;
        t0 = sgn(t0) * powf(fabsf(t0) + 1e-6f, pp) * ws;
        t1 = sgn(t1) * powf(fabsf(t1) + 1e-6f, pp) * ws;
        float acc[HEADS] = {0.f, 0.f, 0.f, 0.f};
        for (int j = 0; j < 512; ++j) {
            float hsum = fmaf(t0, cpb_w1[j], fmaf(t1, cpb_w1[512 + j], cpb_b1[j]));
            hsum = fmaxf(hsum, 0.f);
            #pragma unroll
            for (int h = 0; h < HEADS; ++h) acc[h] += hsum * cpb_w2[j * HEADS + h];
        }
        #pragma unroll
        for (int h = 0; h < HEADS; ++h) table[tid][h] = acc[h];
    }
    __syncthreads();

    for (int e = tid; e < HEADS * NTOK * NTOK; e += blockDim.x) {
        int h = e >> 12, nm = e & 4095;
        biasg[e] = table[rel_index[nm]][h];
    }
    for (int e = tid; e < 384 * DIM; e += blockDim.x) {
        int n = e >> 7, k = e & 127;
        WqkvT[e] = (half_t)W_qkv[k * 384 + n];
    }
    for (int e = tid; e < DIM * DIM; e += blockDim.x) {
        int n = e >> 7, k = e & 127;
        WprojT[e] = (half_t)W_proj[k * DIM + n];
    }
}

// ---- Kernel 1: fully fused window attention (1 block = 1 window) -----------
__global__ void __launch_bounds__(256)
swin_window_attn_kernel(const float* __restrict__ x,
                        const float* __restrict__ b_proj,
                        const float* __restrict__ biasg,
                        const half_t* __restrict__ WqkvT,
                        const half_t* __restrict__ WprojT,
                        float* __restrict__ out) {
    extern __shared__ char smem[];
    half_t* Xh = (half_t*)(smem);            // [64][128] f16 (reused as O)
    half_t* Qh = (half_t*)(smem + 16384);    // [64][128] f16 (pre-scaled)
    half_t* Kh = (half_t*)(smem + 32768);    // [64][128] f16
    half_t* Vt = (half_t*)(smem + 49152);    // [128(hd-major)][64(tok)] f16
    float*  S  = (float*) (smem + 65536);    // [4][64][64] f32 logits
    half_t* P  = (half_t*)(smem + 131072);   // [4][64][64] f16 probs

    const int tid  = threadIdx.x;
    const int lane = tid & 31;
    const int wave = tid >> 5;
    const int n0   = lane & 15;
    const int mb   = (lane & 16) ? 8 : 0;
    const long win = blockIdx.x;
    const float* xw = x + win * (long)(NTOK * DIM);

    // Stage 0: x -> f16 LDS
    for (int i = tid * 4; i < NTOK * DIM; i += 256 * 4) {
        const float4 f = *(const float4*)(xw + i);
        union { half_t h[4]; unsigned long long u; } t;
        t.h[0] = (half_t)f.x; t.h[1] = (half_t)f.y;
        t.h[2] = (half_t)f.z; t.h[3] = (half_t)f.w;
        *(unsigned long long*)(Xh + i) = t.u;
    }
    __syncthreads();

    // Stage 1: QKV = X(64x128) @ WqkvT'(128x384); 96 tiles.
    // mt = t&3 is wave-invariant (stride 8) -> A fragments hoisted once/wave.
    // Two independent accumulator chains overlap the WMMA/B-load latency.
    const float scale = 0.17677669529663687f;  // 32^-0.5 folded into Q
    for (int t = wave; t < 96; t += 8) {
        const int mt = t & 3, nt = t >> 2;
        const half_t* arow = Xh + mt * 16 * DIM;
        const half_t* brow = WqkvT + nt * 16 * DIM;
        v8f acc0 = {}, acc1 = {};
        acc0 = wmma_f16(load_frag_a(arow +  0, DIM, lane), load_frag_b(brow +  0, DIM, lane), acc0);
        acc1 = wmma_f16(load_frag_a(arow + 32, DIM, lane), load_frag_b(brow + 32, DIM, lane), acc1);
        acc0 = wmma_f16(load_frag_a(arow + 64, DIM, lane), load_frag_b(brow + 64, DIM, lane), acc0);
        acc1 = wmma_f16(load_frag_a(arow + 96, DIM, lane), load_frag_b(brow + 96, DIM, lane), acc1);
        v8f acc;
        #pragma unroll
        for (int r = 0; r < 8; ++r) acc[r] = acc0[r] + acc1[r];

        const int gn = nt * 16 + n0;           // 0..383
        if (gn < 128) {                        // Q (scaled)
            #pragma unroll
            for (int r = 0; r < 8; ++r)
                Qh[(mt * 16 + mb + r) * DIM + gn] = (half_t)(acc[r] * scale);
        } else if (gn < 256) {                 // K
            const int kn = gn - 128;
            #pragma unroll
            for (int r = 0; r < 8; ++r)
                Kh[(mt * 16 + mb + r) * DIM + kn] = (half_t)acc[r];
        } else {                               // V stored transposed (hd-major)
            const int vn = gn - 256;           // head*32 + hd
            v8h tv;
            #pragma unroll
            for (int r = 0; r < 8; ++r) tv[r] = (half_t)acc[r];
            *(v8h*)(Vt + vn * NTOK + mt * 16 + mb) = tv;
        }
    }
    __syncthreads();

    // Stage 2: S = Qs @ K^T + bias (bias rides in as the WMMA C operand).
    // t = wave*8+i: head fixed per wave; Q (A) fragment changes every 4 tiles
    // -> unrolled so the compiler CSEs repeated A-fragment LDS loads.
    #pragma unroll
    for (int i = 0; i < 8; ++i) {
        const int t  = wave * 8 + i;
        const int h  = t >> 4, mt = (t >> 2) & 3, nt = t & 3;
        const float* bsrc = biasg + h * 4096 + (mt * 16 + mb) * 64 + nt * 16 + n0;
        v8f c;
        #pragma unroll
        for (int r = 0; r < 8; ++r) c[r] = bsrc[r * 64];
        v16h a = load_frag_a(Qh + mt * 16 * DIM + h * HD, DIM, lane);
        v16h b = load_frag_b(Kh + nt * 16 * DIM + h * HD, DIM, lane);
        v8f d = wmma_f16(a, b, c);
        float* srow = S + h * 4096 + (mt * 16 + mb) * 64 + nt * 16 + n0;
        #pragma unroll
        for (int r = 0; r < 8; ++r) srow[r * 64] = d[r];
    }
    __syncthreads();

    // Stage 3: softmax, one (head,row) per thread. Column index rotated by row
    // so concurrent lanes hit distinct LDS banks (rows are 64 dwords apart).
    {
        const int h = tid >> 6, row = tid & 63;
        float*  srow = S + h * 4096 + row * 64;
        half_t* prow = P + h * 4096 + row * 64;
        float mx = -INFINITY;
        #pragma unroll 8
        for (int j = 0; j < 64; ++j) mx = fmaxf(mx, srow[(j + row) & 63]);
        float sum = 0.f;
        #pragma unroll 8
        for (int j = 0; j < 64; ++j) {
            const int i = (j + row) & 63;
            float e = __expf(srow[i] - mx);
            srow[i] = e;
            sum += e;
        }
        const float inv = 1.f / sum;
        #pragma unroll 8
        for (int j = 0; j < 64; ++j) {
            const int i = (j + row) & 63;
            prow[i] = (half_t)(srow[i] * inv);
        }
    }
    __syncthreads();

    // Stage 4: O = P @ V  (per head: 64x32 out, K=64 -> 2 WMMA per tile).
    // t = wave*4+i: head fixed per wave, P (A) fragments reused across nt.
    half_t* O = Xh;  // X no longer needed
    #pragma unroll
    for (int i = 0; i < 4; ++i) {
        const int t  = wave * 4 + i;
        const int h  = t >> 3, mt = (t >> 1) & 3, nt = t & 1;
        v8f acc = {};
        #pragma unroll
        for (int kk = 0; kk < 2; ++kk) {
            v16h a = load_frag_a(P + h * 4096 + mt * 16 * 64 + kk * 32, 64, lane);
            v16h b = load_frag_b(Vt + (h * HD + nt * 16) * NTOK + kk * 32, NTOK, lane);
            acc = wmma_f16(a, b, acc);
        }
        #pragma unroll
        for (int r = 0; r < 8; ++r)
            O[(mt * 16 + mb + r) * DIM + h * HD + nt * 16 + n0] = (half_t)acc[r];
    }
    __syncthreads();

    // Stage 5: Y = O @ Wproj^T + b_proj, fp32 D straight to HBM.
    // mt = t&3 wave-invariant -> O (A) fragments hoisted; dual acc chains.
    float* yw = out + win * (long)(NTOK * DIM);
    #pragma unroll
    for (int i = 0; i < 4; ++i) {
        const int t  = wave + 8 * i;
        const int mt = t & 3, nt = t >> 2;
        const half_t* arow = O + mt * 16 * DIM;
        const half_t* brow = WprojT + nt * 16 * DIM;
        v8f acc0 = {}, acc1 = {};
        acc0 = wmma_f16(load_frag_a(arow +  0, DIM, lane), load_frag_b(brow +  0, DIM, lane), acc0);
        acc1 = wmma_f16(load_frag_a(arow + 32, DIM, lane), load_frag_b(brow + 32, DIM, lane), acc1);
        acc0 = wmma_f16(load_frag_a(arow + 64, DIM, lane), load_frag_b(brow + 64, DIM, lane), acc0);
        acc1 = wmma_f16(load_frag_a(arow + 96, DIM, lane), load_frag_b(brow + 96, DIM, lane), acc1);
        const float bn = b_proj[nt * 16 + n0];
        float* dst = yw + (mt * 16 + mb) * DIM + nt * 16 + n0;
        #pragma unroll
        for (int r = 0; r < 8; ++r) dst[r * DIM] = acc0[r] + acc1[r] + bn;
    }
}

// ---- Host launch -----------------------------------------------------------
extern "C" void kernel_launch(void* const* d_in, const int* in_sizes, int n_in,
                              void* d_out, int out_size, void* d_ws, size_t ws_size,
                              hipStream_t stream) {
    const float* x        = (const float*)d_in[0];
    const float* W_qkv    = (const float*)d_in[1];
    const float* W_proj   = (const float*)d_in[2];
    const float* b_proj   = (const float*)d_in[3];
    const float* cpb_w1   = (const float*)d_in[4];
    const float* cpb_b1   = (const float*)d_in[5];
    const float* cpb_w2   = (const float*)d_in[6];
    const float* h_times  = (const float*)d_in[7];
    const float* pow_para = (const float*)d_in[8];
    const float* ws1      = (const float*)d_in[9];
    const float* ws2      = (const float*)d_in[10];
    const int*   rel_idx  = (const int*)d_in[11];
    const float* coords   = (const float*)d_in[12];

    // workspace: bias f32 (64KB) | WqkvT f16 (96KB) | WprojT f16 (32KB)
    float*  biasg  = (float*)d_ws;
    half_t* WqkvT  = (half_t*)((char*)d_ws + 65536);
    half_t* WprojT = (half_t*)((char*)d_ws + 65536 + 98304);

    swin_prep_kernel<<<1, 256, 0, stream>>>(W_qkv, W_proj, cpb_w1, cpb_b1, cpb_w2,
                                            h_times, pow_para, ws1, ws2,
                                            rel_idx, coords, biasg, WqkvT, WprojT);

    const size_t smem = 163840;  // 160 KB -> 2 blocks per 320 KB WGP
    (void)hipFuncSetAttribute((const void*)swin_window_attn_kernel,
                              hipFuncAttributeMaxDynamicSharedMemorySize, (int)smem);
    swin_window_attn_kernel<<<4096, 256, smem, stream>>>(x, b_proj, biasg,
                                                         WqkvT, WprojT, (float*)d_out);
}
